// Attention_29738353557815
// MI455X (gfx1250) — compile-verified, matrix operand order checked by reading, not run on hardware
//
#include <hip/hip_runtime.h>
#include <hip/hip_bf16.h>

// ---------------- problem constants ----------------
#define TSEQ 2048
#define DMODEL 2048
#define NHEADS 16
#define NKVH 4
#define NREP 4
#define HDIM 128
#define QCOLS 4096           // NH * HD * 2  (q + gate)
#define KVCOLS 512           // NKV * HD
#define ATT_SCALE 0.08838834764831845f   // 1/sqrt(128)

typedef __attribute__((ext_vector_type(16))) __bf16 v16bf;
typedef __attribute__((ext_vector_type(8)))  __bf16 v8bf;
typedef __attribute__((ext_vector_type(4)))  __bf16 v4bf;
typedef __attribute__((ext_vector_type(8)))  float  v8f;

static __device__ __forceinline__ __bf16 f2bf(float f) {
    unsigned u = __builtin_bit_cast(unsigned, f);
    u += 0x7FFFu + ((u >> 16) & 1u);          // round-to-nearest-even
    unsigned short h = (unsigned short)(u >> 16);
    return __builtin_bit_cast(__bf16, h);
}

static __device__ __forceinline__ v16bf combine8(v8bf lo, v8bf hi) {
    v16bf r;
#pragma unroll
    for (int i = 0; i < 8; ++i) { r[i] = lo[i]; r[i + 8] = hi[i]; }
    return r;
}

static __device__ __forceinline__ v8f wmma_bf16(v16bf a, v16bf b, v8f c) {
    return __builtin_amdgcn_wmma_f32_16x16x32_bf16(
        false, a, false, b, (short)0, c, false, false);
}

// async 16-byte global -> LDS copy (per-lane), tracked by ASYNCcnt
static __device__ __forceinline__ void async_copy_b128(const __bf16* g, const __bf16* l) {
    unsigned long long ga = (unsigned long long)(uintptr_t)g;
    unsigned la = (unsigned)(uintptr_t)l;     // low 32 bits == LDS offset
    asm volatile("global_load_async_to_lds_b128 %0, %1, off"
                 :: "v"(la), "v"(ga) : "memory");
}
static __device__ __forceinline__ void wait_async0() {
    asm volatile("s_wait_asynccnt 0x0" ::: "memory");
}

// =====================================================================
// f32 -> bf16 bulk convert (row-major, for the activation matrix)
// =====================================================================
__global__ __launch_bounds__(256) void cvt_bf16_kernel(
    const float* __restrict__ src, __bf16* __restrict__ dst, int n4)
{
    int i = blockIdx.x * blockDim.x + threadIdx.x;
    if (i < n4) {
        float4 v = ((const float4*)src)[i];
        v4bf o; o[0] = f2bf(v.x); o[1] = f2bf(v.y); o[2] = f2bf(v.z); o[3] = f2bf(v.w);
        ((v4bf*)dst)[i] = o;
    }
}

// =====================================================================
// f32[K][N] -> bf16[N][K] transpose-convert (one-shot, for weights) so
// GEMM B-tiles become contiguous rows -> pure async LDS staging.
// 32x32 tiles, block (32,8).
// =====================================================================
__global__ __launch_bounds__(256) void cvt_t_bf16_kernel(
    const float* __restrict__ src, __bf16* __restrict__ dst, int Kd, int Nd)
{
    __shared__ float tile[32][33];
    const int tx = threadIdx.x, ty = threadIdx.y;
    const int n0 = blockIdx.x * 32, k0 = blockIdx.y * 32;
#pragma unroll
    for (int j = 0; j < 4; ++j)
        tile[ty + j * 8][tx] = src[(size_t)(k0 + ty + j * 8) * Nd + n0 + tx];
    __syncthreads();
#pragma unroll
    for (int j = 0; j < 4; ++j)
        dst[(size_t)(n0 + ty + j * 8) * Kd + k0 + tx] = f2bf(tile[tx][ty + j * 8]);
}

// =====================================================================
// bf16 [R][C] -> [C][R] transpose per head (for V -> V^T), block (32,8)
// grid (C/32, R/32, NKVH)
// =====================================================================
__global__ __launch_bounds__(256) void tr_bf16_kernel(
    const __bf16* __restrict__ src, __bf16* __restrict__ dst, int R, int C)
{
    __shared__ __bf16 tile[32][33];
    const int tx = threadIdx.x, ty = threadIdx.y;
    const int c0 = blockIdx.x * 32, r0 = blockIdx.y * 32;
    const size_t hb = (size_t)blockIdx.z * R * C;
#pragma unroll
    for (int j = 0; j < 4; ++j)
        tile[ty + j * 8][tx] = src[hb + (size_t)(r0 + ty + j * 8) * C + c0 + tx];
    __syncthreads();
#pragma unroll
    for (int j = 0; j < 4; ++j)
        dst[hb + (size_t)(c0 + ty + j * 8) * R + r0 + tx] = tile[tx][ty + j * 8];
}

// =====================================================================
// GEMM: C[M,N](f32) = A[M,K](bf16, row-major) * Bt[N,K](bf16, pre-
// transposed).  Block 128x128, BK=32, 256 thr = 8 waves, wave 32x64.
// Both tiles async-staged (global_load_async_to_lds_b128), double-
// buffered LDS so next-tile staging overlaps WMMA.
// =====================================================================
#define LDA_S 40   // 32 + 8 bf16 pad -> 80B row stride (16B-aligned chunks)

template<int N, int K>
__global__ __launch_bounds__(256) void gemm_bf16_kernel(
    const __bf16* __restrict__ A, const __bf16* __restrict__ Bt,
    float* __restrict__ C)
{
    __shared__ __bf16 As[2][128 * LDA_S];   // [m][k]
    __shared__ __bf16 Bs[2][128 * LDA_S];   // [n][k]

    const int m0 = blockIdx.y * 128;
    const int n0 = blockIdx.x * 128;
    const int tid  = threadIdx.x;
    const int wid  = tid >> 5;
    const int lane = tid & 31;
    const int m16  = lane & 15;
    const int g    = lane >> 4;
    const int wm   = wid & 3;     // 4 waves along M
    const int wn   = wid >> 2;    // 2 waves along N

    auto stage = [&](int buf, int k0) {
#pragma unroll
        for (int i = 0; i < 2; ++i) {
            int idx = tid + i * 256;                    // 512 chunks per tile
            int row = idx >> 2, c = idx & 3;
            async_copy_b128(A  + (size_t)(m0 + row) * K + k0 + c * 8,
                            &As[buf][row * LDA_S + c * 8]);
            async_copy_b128(Bt + (size_t)(n0 + row) * K + k0 + c * 8,
                            &Bs[buf][row * LDA_S + c * 8]);
        }
    };

    v8f acc[2][4];
#pragma unroll
    for (int mi = 0; mi < 2; ++mi)
#pragma unroll
        for (int ni = 0; ni < 4; ++ni) acc[mi][ni] = (v8f)0.0f;

    stage(0, 0);
    int buf = 0;
    for (int k0 = 0; k0 < K; k0 += 32) {
        wait_async0();
        __syncthreads();
        if (k0 + 32 < K) stage(buf ^ 1, k0 + 32);

        v16bf af[2], bfr[4];
#pragma unroll
        for (int mi = 0; mi < 2; ++mi) {
            const __bf16* p = &As[buf][(wm * 32 + mi * 16 + m16) * LDA_S + g * 8];
            af[mi] = combine8(*(const v8bf*)p, *(const v8bf*)(p + 16));
        }
#pragma unroll
        for (int ni = 0; ni < 4; ++ni) {
            const __bf16* p = &Bs[buf][(wn * 64 + ni * 16 + m16) * LDA_S + g * 8];
            bfr[ni] = combine8(*(const v8bf*)p, *(const v8bf*)(p + 16));
        }
#pragma unroll
        for (int mi = 0; mi < 2; ++mi)
#pragma unroll
            for (int ni = 0; ni < 4; ++ni)
                acc[mi][ni] = wmma_bf16(af[mi], bfr[ni], acc[mi][ni]);
        buf ^= 1;
    }

    // ---- epilogue: C/D layout -> row m = r + 8*g, col n = lane&15 ----
#pragma unroll
    for (int mi = 0; mi < 2; ++mi)
#pragma unroll
        for (int ni = 0; ni < 4; ++ni) {
            float* cp = C + (size_t)(m0 + wm * 32 + mi * 16 + 8 * g) * N
                          + n0 + wn * 64 + ni * 16 + m16;
#pragma unroll
            for (int r = 0; r < 8; ++r) cp[(size_t)r * N] = acc[mi][ni][r];
        }
}

// =====================================================================
// RMS-norm + RoPE + bf16 pack.  block = 128 threads = one (t, head).
// grid.y: 0..15 -> q heads, 16..19 -> k heads, 20..23 -> v heads.
// =====================================================================
__global__ __launch_bounds__(128) void norm_rope_kernel(
    const float* __restrict__ qout, const float* __restrict__ kout,
    const float* __restrict__ vout, const float* __restrict__ cosb,
    const float* __restrict__ sinb, const float* __restrict__ qw,
    const float* __restrict__ kw,
    __bf16* __restrict__ qb, __bf16* __restrict__ kb, __bf16* __restrict__ vb)
{
    const int t = blockIdx.x;
    const int y = blockIdx.y;
    const int i = threadIdx.x;

    __shared__ float ssq[128];
    __shared__ float sxn[128];

    float x; const float* w; __bf16* dst;
    if (y < NHEADS) {                       // q
        x = qout[(size_t)t * QCOLS + y * (2 * HDIM) + i];
        w = qw; dst = qb + ((size_t)y * TSEQ + t) * HDIM;
    } else if (y < NHEADS + NKVH) {         // k
        int h = y - NHEADS;
        x = kout[(size_t)t * KVCOLS + h * HDIM + i];
        w = kw; dst = kb + ((size_t)h * TSEQ + t) * HDIM;
    } else {                                // v: plain bf16 pack
        int h = y - NHEADS - NKVH;
        x = vout[(size_t)t * KVCOLS + h * HDIM + i];
        vb[((size_t)h * TSEQ + t) * HDIM + i] = f2bf(x);
        return;
    }

    ssq[i] = x * x;
    __syncthreads();
#pragma unroll
    for (int s = 64; s > 0; s >>= 1) {
        if (i < s) ssq[i] += ssq[i + s];
        __syncthreads();
    }
    const float r = rsqrtf(ssq[0] * (1.0f / HDIM) + 1e-6f);
    const float xn = x * r * w[i];
    sxn[i] = xn;
    __syncthreads();
    const float rot = (i < 64) ? -sxn[i + 64] : sxn[i - 64];
    const float val = xn * cosb[t * HDIM + i] + rot * sinb[t * HDIM + i];
    dst[i] = f2bf(val);
}

// =====================================================================
// Flash attention with WMMA.  block = 128 thr = 4 waves, 64 q-rows/block,
// grid = (T/64, NH).  Q fragments VGPR-resident; K and V^T tiles fully
// async-staged into double-buffered LDS; P relayout via per-wave LDS;
// online softmax; sigmoid-gate epilogue writes bf16 attn[T][NH*HD].
// =====================================================================
#define KT_S 136   // 128 + 8 pad (bf16)
#define VT_S 72    // 64 + 8 pad

__global__ __launch_bounds__(128) void flash_attn_kernel(
    const __bf16* __restrict__ qb, const __bf16* __restrict__ kb,
    const __bf16* __restrict__ vtb, const float* __restrict__ qout,
    const int* __restrict__ seg, __bf16* __restrict__ attn)
{
    __shared__ __bf16 Kt[2][64 * KT_S];     // [key][hd]
    __shared__ __bf16 Vt[2][128 * VT_S];    // [hd][key]
    __shared__ __bf16 Pt[4][16 * 64];       // per-wave P relayout

    const int q0   = blockIdx.x * 64;
    const int head = blockIdx.y;
    const int kvh  = head / NREP;
    const int tid  = threadIdx.x;
    const int wid  = tid >> 5;
    const int lane = tid & 31;
    const int m16  = lane & 15;
    const int g    = lane >> 4;
    const int qrow = q0 + wid * 16;

    auto stage = [&](int buf, int kt0) {
#pragma unroll
        for (int i = 0; i < 8; ++i) {
            int idx = tid + i * 128;               // 1024 chunks each
            int krow = idx >> 4, kch = idx & 15;   // K tile 64x128
            async_copy_b128(kb + ((size_t)kvh * TSEQ + kt0 + krow) * HDIM + kch * 8,
                            &Kt[buf][krow * KT_S + kch * 8]);
            int vrow = idx >> 3, vch = idx & 7;    // V^T tile 128x64
            async_copy_b128(vtb + ((size_t)kvh * HDIM + vrow) * TSEQ + kt0 + vch * 8,
                            &Vt[buf][vrow * VT_S + vch * 8]);
        }
    };

    // Q fragments: 4 k-steps of 32 over HD=128, resident in VGPRs
    v16bf qf[4];
#pragma unroll
    for (int ks = 0; ks < 4; ++ks) {
        const __bf16* p = qb + ((size_t)head * TSEQ + qrow + m16) * HDIM + ks * 32 + g * 8;
        qf[ks] = combine8(*(const v8bf*)p, *(const v8bf*)(p + 16));
    }

    int segq[8];
#pragma unroll
    for (int r = 0; r < 8; ++r) segq[r] = seg[qrow + 8 * g + r];

    v8f o[8];
#pragma unroll
    for (int no = 0; no < 8; ++no) o[no] = (v8f)0.0f;
    float mrow[8], lrow[8];
#pragma unroll
    for (int r = 0; r < 8; ++r) { mrow[r] = -3.0e38f; lrow[r] = 0.0f; }

    stage(0, 0);
    int buf = 0;
    for (int kt0 = 0; kt0 <= q0; kt0 += 64) {
        wait_async0();
        __syncthreads();
        if (kt0 + 64 <= q0) stage(buf ^ 1, kt0 + 64);

        // ---- S = Q K^T  (4 tiles of 16x16, each 4 wmma k-steps) ----
        float s[4][8];
#pragma unroll
        for (int ni = 0; ni < 4; ++ni) {
            v8f acc = (v8f)0.0f;
#pragma unroll
            for (int ks = 0; ks < 4; ++ks) {
                const __bf16* p = &Kt[buf][(ni * 16 + m16) * KT_S + ks * 32 + g * 8];
                acc = wmma_bf16(qf[ks], combine8(*(const v8bf*)p, *(const v8bf*)(p + 16)), acc);
            }
            const int kidx = kt0 + ni * 16 + m16;
            const int sk = seg[kidx];
#pragma unroll
            for (int r = 0; r < 8; ++r) {
                const int qi = qrow + 8 * g + r;
                const bool ok = (kidx <= qi) && (sk == segq[r]);
                s[ni][r] = ok ? acc[r] * ATT_SCALE : -3.0e38f;
            }
        }

        // ---- online softmax update ----
#pragma unroll
        for (int r = 0; r < 8; ++r) {
            float rm = s[0][r];
#pragma unroll
            for (int ni = 1; ni < 4; ++ni) rm = fmaxf(rm, s[ni][r]);
            rm = fmaxf(rm, __shfl_xor(rm, 8));
            rm = fmaxf(rm, __shfl_xor(rm, 4));
            rm = fmaxf(rm, __shfl_xor(rm, 2));
            rm = fmaxf(rm, __shfl_xor(rm, 1));
            const float mnew  = fmaxf(mrow[r], rm);
            const float alpha = __expf(mrow[r] - mnew);
            float rs = 0.0f;
#pragma unroll
            for (int ni = 0; ni < 4; ++ni) {
                float p = (s[ni][r] <= -1.0e37f) ? 0.0f : __expf(s[ni][r] - mnew);
                s[ni][r] = p; rs += p;
            }
            rs += __shfl_xor(rs, 8);
            rs += __shfl_xor(rs, 4);
            rs += __shfl_xor(rs, 2);
            rs += __shfl_xor(rs, 1);
            lrow[r] = lrow[r] * alpha + rs;
            mrow[r] = mnew;
#pragma unroll
            for (int no = 0; no < 8; ++no) o[no][r] *= alpha;
        }

        // ---- relayout P (C-layout -> A-fragment) via per-wave LDS ----
#pragma unroll
        for (int ni = 0; ni < 4; ++ni)
#pragma unroll
            for (int r = 0; r < 8; ++r)
                Pt[wid][(r + 8 * g) * 64 + ni * 16 + m16] = f2bf(s[ni][r]);
        asm volatile("" ::: "memory");     // DS pipe is in-order within a wave
        v16bf pf[2];
#pragma unroll
        for (int ks = 0; ks < 2; ++ks) {
            const __bf16* p = &Pt[wid][m16 * 64 + ks * 32 + g * 8];
            pf[ks] = combine8(*(const v8bf*)p, *(const v8bf*)(p + 16));
        }

        // ---- O += P V  (8 out tiles x 2 k-steps) ----
#pragma unroll
        for (int no = 0; no < 8; ++no)
#pragma unroll
            for (int ks = 0; ks < 2; ++ks) {
                const __bf16* p = &Vt[buf][(no * 16 + m16) * VT_S + ks * 32 + g * 8];
                o[no] = wmma_bf16(pf[ks], combine8(*(const v8bf*)p, *(const v8bf*)(p + 16)), o[no]);
            }
        buf ^= 1;
    }

    // ---- epilogue: 1/l + sigmoid gate, write bf16 attn[T][NH*HD] ----
#pragma unroll
    for (int r = 0; r < 8; ++r) {
        const int row = qrow + 8 * g + r;
        const float inv = (lrow[r] > 0.0f) ? 1.0f / lrow[r] : 0.0f;
#pragma unroll
        for (int no = 0; no < 8; ++no) {
            const int col = no * 16 + m16;
            const float gate = qout[(size_t)row * QCOLS + head * (2 * HDIM) + HDIM + col];
            const float sg = 1.0f / (1.0f + __expf(-gate));
            attn[(size_t)row * DMODEL + head * HDIM + col] = f2bf(o[no][r] * inv * sg);
        }
    }
}

// =====================================================================
// launcher
// =====================================================================
extern "C" void kernel_launch(void* const* d_in, const int* in_sizes, int n_in,
                              void* d_out, int out_size, void* d_ws, size_t ws_size,
                              hipStream_t stream) {
    (void)in_sizes; (void)n_in; (void)out_size; (void)ws_size;
    const float* hidden = (const float*)d_in[0];
    const float* cosb   = (const float*)d_in[1];
    const float* sinb   = (const float*)d_in[2];
    const int*   seg    = (const int*)d_in[3];
    /* d_in[4] position_ids == arange -> implicit */
    const float* wq     = (const float*)d_in[5];
    const float* wk     = (const float*)d_in[6];
    const float* wv     = (const float*)d_in[7];
    const float* wo     = (const float*)d_in[8];
    const float* qw     = (const float*)d_in[9];
    const float* kw     = (const float*)d_in[10];

    char* ws = (char*)d_ws;
    __bf16* hid_bf = (__bf16*)ws; ws += (size_t)TSEQ * DMODEL * 2;             //  8 MiB
    __bf16* wqT    = (__bf16*)ws; ws += (size_t)DMODEL * QCOLS * 2;            // 16 MiB
    __bf16* wkT    = (__bf16*)ws; ws += (size_t)DMODEL * KVCOLS * 2;           //  2 MiB
    __bf16* wvT    = (__bf16*)ws; ws += (size_t)DMODEL * KVCOLS * 2;           //  2 MiB
    __bf16* woT    = (__bf16*)ws; ws += (size_t)DMODEL * DMODEL * 2;           //  8 MiB
    float* q_out = (float*)ws;  ws += (size_t)TSEQ * QCOLS * sizeof(float);    // 32 MiB
    float* k_out = (float*)ws;  ws += (size_t)TSEQ * KVCOLS * sizeof(float);   //  4 MiB
    float* v_out = (float*)ws;  ws += (size_t)TSEQ * KVCOLS * sizeof(float);   //  4 MiB
    __bf16* q_bf = (__bf16*)ws; ws += (size_t)NHEADS * TSEQ * HDIM * 2;        //  8 MiB
    __bf16* k_bf = (__bf16*)ws; ws += (size_t)NKVH * TSEQ * HDIM * 2;          //  2 MiB
    __bf16* v_bf = (__bf16*)ws; ws += (size_t)NKVH * TSEQ * HDIM * 2;          //  2 MiB
    __bf16* vT_bf = (__bf16*)ws; ws += (size_t)NKVH * TSEQ * HDIM * 2;         //  2 MiB
    __bf16* attnb = (__bf16*)ws; ws += (size_t)TSEQ * DMODEL * 2;              //  8 MiB

    // 0) one-shot conversions: hidden row-major; weights transposed [N][K]
    {
        int n4 = (int)((size_t)TSEQ * DMODEL / 4);
        cvt_bf16_kernel<<<(n4 + 255) / 256, 256, 0, stream>>>(hidden, hid_bf, n4);
    }
    cvt_t_bf16_kernel<<<dim3(QCOLS / 32, DMODEL / 32), dim3(32, 8), 0, stream>>>(
        wq, wqT, DMODEL, QCOLS);
    cvt_t_bf16_kernel<<<dim3(KVCOLS / 32, DMODEL / 32), dim3(32, 8), 0, stream>>>(
        wk, wkT, DMODEL, KVCOLS);
    cvt_t_bf16_kernel<<<dim3(KVCOLS / 32, DMODEL / 32), dim3(32, 8), 0, stream>>>(
        wv, wvT, DMODEL, KVCOLS);
    cvt_t_bf16_kernel<<<dim3(DMODEL / 32, DMODEL / 32), dim3(32, 8), 0, stream>>>(
        wo, woT, DMODEL, DMODEL);

    // 1) projections (fully async-staged, double-buffered WMMA GEMMs)
    gemm_bf16_kernel<QCOLS, DMODEL><<<dim3(QCOLS / 128, TSEQ / 128), 256, 0, stream>>>(
        hid_bf, wqT, q_out);
    gemm_bf16_kernel<KVCOLS, DMODEL><<<dim3(KVCOLS / 128, TSEQ / 128), 256, 0, stream>>>(
        hid_bf, wkT, k_out);
    gemm_bf16_kernel<KVCOLS, DMODEL><<<dim3(KVCOLS / 128, TSEQ / 128), 256, 0, stream>>>(
        hid_bf, wvT, v_out);

    // 2) RMS-norm + RoPE + bf16 pack, then V -> V^T per kv head
    norm_rope_kernel<<<dim3(TSEQ, NHEADS + 2 * NKVH), 128, 0, stream>>>(
        q_out, k_out, v_out, cosb, sinb, qw, kw, q_bf, k_bf, v_bf);
    tr_bf16_kernel<<<dim3(HDIM / 32, TSEQ / 32, NKVH), dim3(32, 8), 0, stream>>>(
        v_bf, vT_bf, TSEQ, HDIM);

    // 3) flash attention + gate (bf16 out)
    flash_attn_kernel<<<dim3(TSEQ / 64, NHEADS), 128, 0, stream>>>(
        q_bf, k_bf, vT_bf, q_out, seg, attnb);

    // 4) output projection
    gemm_bf16_kernel<DMODEL, DMODEL><<<dim3(DMODEL / 128, TSEQ / 128), 256, 0, stream>>>(
        attnb, woT, (float*)d_out);
}
